// CDMamba_seg_cd_5145370821211
// MI455X (gfx1250) — matrix operand order repaired and checked by reading, compiled, not run on hardware
//
#include <hip/hip_runtime.h>
#include <hip/hip_bf16.h>
#include <math.h>

// ---------------------------------------------------------------------------
// ModifiedSRCMLayer for MI455X (gfx1250, wave32, WMMA + async-LDS streaming)
//
//   B=4, C=128, H=W=64, L=4096, M=B*L=16384
//   G=4 groups, DM=32, DI=64, DS=16, DC=4, DTR=2, OUT=128
// ---------------------------------------------------------------------------

#define BSZ   4
#define CCH   128
#define HH    64
#define WW    64
#define LSEQ  4096
#define MTOK  16384     // BSZ * LSEQ
#define NG    4
#define DMG   32
#define DIN   64
#define DST   16
#define DCV   4
#define DTR   2
#define NOUT  128
#define DBLW  48        // padded width for (DTR + 2*DS)=34
#define EPSV  1e-5f
#define SCH   16        // scan chunk length (steps per async LDS chunk)

typedef __attribute__((ext_vector_type(16))) _Float16 v16h;
typedef __attribute__((ext_vector_type(8)))  float    v8f;

// --- gfx1250 async global->LDS path (guarded; fallback = plain LDS fill) ---
#if defined(__has_builtin)
#if __has_builtin(__builtin_amdgcn_global_load_async_to_lds_b128) && \
    __has_builtin(__builtin_amdgcn_s_wait_asynccnt)
#define HAVE_ASYNC_LDS 1
#endif
#endif

#ifdef HAVE_ASYNC_LDS
// Probe-discovered parameter type: pointer to 16-byte int vector, generic AS.
typedef __attribute__((__vector_size__(4 * sizeof(int)))) int v4i_t;
#define GPTR(p) ((v4i_t*)(p))
#define LPTR(p) ((v4i_t*)(p))
#endif

// ---------------------------------------------------------------------------
// WMMA fragment helpers (layouts per CDNA5 ISA 7.12.2, wave32)
// ---------------------------------------------------------------------------
__device__ __forceinline__ v16h ldA(const float* A, int lda, int lane) {
  v16h a;
  int m  = lane & 15;
  int kh = (lane >> 4) << 3;          // 0 or 8
#pragma unroll
  for (int i = 0; i < 8; ++i) {
    int kb = ((i < 4) ? (2 * i) : (16 + 2 * (i - 4))) + kh;
    a[2 * i]     = (_Float16)A[(size_t)m * lda + kb];
    a[2 * i + 1] = (_Float16)A[(size_t)m * lda + kb + 1];
  }
  return a;
}

// A fragment from f16 LDS tile with row stride CCH
__device__ __forceinline__ v16h ldA_sh(const _Float16* A, int k0, int lane) {
  v16h a;
  int m  = lane & 15;
  int kh = (lane >> 4) << 3;
#pragma unroll
  for (int i = 0; i < 8; ++i) {
    int kb = ((i < 4) ? (2 * i) : (16 + 2 * (i - 4))) + kh + k0;
    a[2 * i]     = A[m * CCH + kb];
    a[2 * i + 1] = A[m * CCH + kb + 1];
  }
  return a;
}

// A fragment summing two f32 sources (yf + yb)
__device__ __forceinline__ v16h ldA2(const float* A0, const float* A1, int lda, int lane) {
  v16h a;
  int m  = lane & 15;
  int kh = (lane >> 4) << 3;
#pragma unroll
  for (int i = 0; i < 8; ++i) {
    int kb = ((i < 4) ? (2 * i) : (16 + 2 * (i - 4))) + kh;
    size_t o0 = (size_t)m * lda + kb;
    a[2 * i]     = (_Float16)(A0[o0] + A1[o0]);
    a[2 * i + 1] = (_Float16)(A0[o0 + 1] + A1[o0 + 1]);
  }
  return a;
}

// W row-major (N x K) -> B^T fragment; nrem = valid rows past n0
__device__ __forceinline__ v16h ldBT(const float* W, int ldw, int nrem, int lane) {
  v16h b;
  int n  = lane & 15;
  int kh = (lane >> 4) << 3;          // 0 or 8
  bool ok = (n < nrem);
#pragma unroll
  for (int i = 0; i < 8; ++i) {
    int k = ((i < 4) ? (2 * i) : (16 + 2 * (i - 4))) + kh;
    b[2 * i]     = ok ? (_Float16)W[(size_t)n * ldw + k]     : (_Float16)0.0f;
    b[2 * i + 1] = ok ? (_Float16)W[(size_t)n * ldw + k + 1] : (_Float16)0.0f;
  }
  return b;
}

#define WMMA_F32_F16(a, b, c) \
  __builtin_amdgcn_wmma_f32_16x16x32_f16(false, (a), false, (b), (short)0, (c), false, false)

__device__ __forceinline__ float silu_f(float v) { return v / (1.0f + __expf(-v)); }

// ---------------------------------------------------------------------------
// K1: ConvPosEnc + pos_embed (identity resize) + LayerNorm.
// ---------------------------------------------------------------------------
__global__ void k_prep(const float* __restrict__ x,
                       const float* __restrict__ pcw,
                       const float* __restrict__ pcb,
                       const float* __restrict__ pemb,
                       const float* __restrict__ lng,
                       const float* __restrict__ lnb,
                       float* __restrict__ xs, float* __restrict__ xn) {
  int tok = blockIdx.x;               // b*L + l
  int c   = threadIdx.x;              // 0..127
  int b   = tok >> 12;
  int l   = tok & (LSEQ - 1);
  int h   = l >> 6, w = l & 63;

  const float* xp = x + (((size_t)b * CCH + c) * HH) * WW;
  float conv = 0.0f;
#pragma unroll
  for (int dy = -1; dy <= 1; ++dy) {
#pragma unroll
    for (int dx = -1; dx <= 1; ++dx) {
      int hh = h + dy, ww = w + dx;
      if (hh >= 0 && hh < HH && ww >= 0 && ww < WW)
        conv += xp[hh * WW + ww] * pcw[c * 9 + (dy + 1) * 3 + (dx + 1)];
    }
  }
  float v = xp[h * WW + w] + conv + pcb[c] + pemb[(size_t)l * CCH + c];

  __shared__ float s1[CCH], s2[CCH];
  s1[c] = v; s2[c] = v * v;
  __syncthreads();
  for (int s = CCH / 2; s > 0; s >>= 1) {
    if (c < s) { s1[c] += s1[c + s]; s2[c] += s2[c + s]; }
    __syncthreads();
  }
  float mu  = s1[0] * (1.0f / CCH);
  float var = s2[0] * (1.0f / CCH) - mu * mu;
  float r   = rsqrtf(var + EPSV);

  xs[(size_t)tok * CCH + c] = v;
  xn[(size_t)tok * CCH + c] = (v - mu) * r * lng[c] + lnb[c];
}

// ---------------------------------------------------------------------------
// K2: xz = xn_g @ Win_g^T. K=32 -> one WMMA. grid(M/16, 8*G), block 32.
// ---------------------------------------------------------------------------
__global__ void k_inproj(const float* __restrict__ xn,
                         const float* __restrict__ Win,
                         float* __restrict__ u, float* __restrict__ z) {
  int lane = threadIdx.x;
  int m0   = blockIdx.x * 16;
  int nt   = blockIdx.y & 7;
  int g    = blockIdx.y >> 3;
  int n0   = nt * 16;

  const float* Ap = xn + (size_t)m0 * CCH + g * DMG;
  const float* Wp = Win + ((size_t)g * 2 * DIN + n0) * DMG;

  v16h a = ldA(Ap, CCH, lane);
  v16h b = ldBT(Wp, DMG, 16, lane);
  v8f  c = {};
  c = WMMA_F32_F16(a, b, c);

  int n = n0 + (lane & 15);
  float* dst = (n < DIN) ? (u + ((size_t)g * MTOK) * DIN + n)
                         : (z + ((size_t)g * MTOK) * DIN + (n - DIN));
#pragma unroll
  for (int r = 0; r < 8; ++r) {
    int m = m0 + r + 8 * (lane >> 4);
    dst[(size_t)m * DIN] = c[r];
  }
}

// ---------------------------------------------------------------------------
// K3: causal depthwise conv (DC=4) along scan order + SiLU.
// ---------------------------------------------------------------------------
__global__ void k_dwconv(const float* __restrict__ u,
                         const float* __restrict__ cw,
                         const float* __restrict__ cb,
                         float* __restrict__ uc) {
  size_t idx = (size_t)blockIdx.x * blockDim.x + threadIdx.x;
  int di = idx & (DIN - 1);
  size_t r = idx >> 6;
  int t  = r & (LSEQ - 1); r >>= 12;
  int b  = r & 3;          r >>= 2;
  int d  = r & 1;          r >>= 1;
  int g  = (int)r;

  const float* up = u + ((size_t)g * MTOK + (size_t)b * LSEQ) * DIN + di;
  const float* wp = cw + (((size_t)(g * 2 + d)) * DIN + di) * DCV;
  float acc = cb[(size_t)(g * 2 + d) * DIN + di];
#pragma unroll
  for (int j = 0; j < DCV; ++j) {
    int tt = t - (DCV - 1) + j;
    if (tt >= 0) {
      int sl = (d == 0) ? tt : (LSEQ - 1 - tt);
      acc += wp[j] * up[(size_t)sl * DIN];
    }
  }
  uc[((size_t)(g * 2 + d) * MTOK + (size_t)b * LSEQ + t) * DIN + di] = silu_f(acc);
}

// ---------------------------------------------------------------------------
// K4: dbl = uc @ xproj_W^T. K=64 (2 WMMA), N=34 padded to 48.
// ---------------------------------------------------------------------------
__global__ void k_xproj(const float* __restrict__ uc,
                        const float* __restrict__ xw,
                        float* __restrict__ dbl) {
  int lane = threadIdx.x;
  int m0   = blockIdx.x * 16;
  int nt   = blockIdx.y % 3;
  int d    = (blockIdx.y / 3) & 1;
  int g    = blockIdx.y / 6;
  int n0   = nt * 16;
  int gd   = g * 2 + d;

  const float* Ap = uc + ((size_t)gd * MTOK + m0) * DIN;
  const float* Wp = xw + ((size_t)gd * (DTR + 2 * DST) + n0) * DIN;
  int nrem = (DTR + 2 * DST) - n0;

  v8f c = {};
#pragma unroll
  for (int k0 = 0; k0 < DIN; k0 += 32) {
    v16h a = ldA(Ap + k0, DIN, lane);
    v16h b = ldBT(Wp + k0, DIN, nrem, lane);
    c = WMMA_F32_F16(a, b, c);
  }

  int n = n0 + (lane & 15);
  if (n < DTR + 2 * DST) {
    float* dst = dbl + ((size_t)gd * MTOK) * DBLW + n;
#pragma unroll
    for (int r = 0; r < 8; ++r) {
      int m = m0 + r + 8 * (lane >> 4);
      dst[(size_t)m * DBLW] = c[r];
    }
  }
}

// ---------------------------------------------------------------------------
// K5: dt = softplus(dbl[:, :2] @ dt_W^T + dt_b).
// ---------------------------------------------------------------------------
__global__ void k_dt(const float* __restrict__ dbl,
                     const float* __restrict__ dtw,
                     const float* __restrict__ dtb,
                     float* __restrict__ dt) {
  size_t idx = (size_t)blockIdx.x * blockDim.x + threadIdx.x;
  int di = idx & (DIN - 1);
  size_t r = idx >> 6;
  int m  = r & (MTOK - 1);
  int gd = (int)(r >> 14);

  const float* dp = dbl + ((size_t)gd * MTOK + m) * DBLW;
  float v = dp[0] * dtw[((size_t)gd * DIN + di) * 2 + 0]
          + dp[1] * dtw[((size_t)gd * DIN + di) * 2 + 1]
          + dtb[(size_t)gd * DIN + di];
  float sp = (v > 20.0f) ? v : log1pf(__expf(v));
  dt[((size_t)gd * MTOK + m) * DIN + di] = sp;
}

// ---------------------------------------------------------------------------
// K6: selective scan with double-buffered async global->LDS streaming.
// One block per (g,d,b) = 32 blocks, 256 threads.
// Chunk of SCH=16 steps: uc 4096B + dt 4096B + dbl 3072B, contiguous rows ->
// one global_load_async_to_lds_b128 per lane, retired by s_wait_asynccnt 0.
// ---------------------------------------------------------------------------
__device__ __forceinline__ void scan_issue_chunk(const float* gu, const float* gt,
                                                 const float* gb,
                                                 float* lu, float* lt, float* lb,
                                                 int tid) {
#ifdef HAVE_ASYNC_LDS
  __builtin_amdgcn_global_load_async_to_lds_b128(GPTR(gu + tid * 4), LPTR(lu + tid * 4), 0, 0);
  __builtin_amdgcn_global_load_async_to_lds_b128(GPTR(gt + tid * 4), LPTR(lt + tid * 4), 0, 0);
  if (tid < (SCH * DBLW) / 4)
    __builtin_amdgcn_global_load_async_to_lds_b128(GPTR(gb + tid * 4), LPTR(lb + tid * 4), 0, 0);
#else
  for (int i = tid; i < SCH * DIN; i += 256) { lu[i] = gu[i]; lt[i] = gt[i]; }
  for (int i = tid; i < SCH * DBLW; i += 256) lb[i] = gb[i];
#endif
}

__device__ __forceinline__ void scan_wait_chunk() {
#ifdef HAVE_ASYNC_LDS
  __builtin_amdgcn_s_wait_asynccnt(0);
#endif
  __syncthreads();
}

__global__ void k_scan(const float* __restrict__ uc,
                       const float* __restrict__ dt,
                       const float* __restrict__ dbl,
                       const float* __restrict__ z,
                       const float* __restrict__ Alog,
                       const float* __restrict__ Dsk,
                       float* __restrict__ yf, float* __restrict__ yb) {
  __shared__ float s_uc[2][SCH * DIN];
  __shared__ float s_dt[2][SCH * DIN];
  __shared__ float s_db[2][SCH * DBLW];

  int g = blockIdx.x >> 3;
  int d = (blockIdx.x >> 2) & 1;
  int b = blockIdx.x & 3;
  int gd = g * 2 + d;

  int t   = threadIdx.x;              // 0..255
  int di  = t >> 2;
  int dsq = t & 3;

  const float* ucp  = uc  + ((size_t)gd * MTOK + (size_t)b * LSEQ) * DIN;
  const float* dtp  = dt  + ((size_t)gd * MTOK + (size_t)b * LSEQ) * DIN;
  const float* dblp = dbl + ((size_t)gd * MTOK + (size_t)b * LSEQ) * DBLW;
  const float* zp   = z   + ((size_t)g * MTOK + (size_t)b * LSEQ) * DIN;
  float* yout = ((d == 0) ? yf : yb) + ((size_t)g * MTOK + (size_t)b * LSEQ) * DIN;

  float A[4];
#pragma unroll
  for (int i = 0; i < 4; ++i)
    A[i] = -__expf(Alog[(((size_t)gd * DIN + di) * DST) + dsq * 4 + i]);
  float Dv = Dsk[(size_t)gd * DIN + di];

  float h[4] = {0.f, 0.f, 0.f, 0.f};

  const int NC = LSEQ / SCH;          // 256 chunks
  scan_issue_chunk(ucp, dtp, dblp, s_uc[0], s_dt[0], s_db[0], t);

  for (int c = 0; c < NC; ++c) {
    scan_wait_chunk();                // chunk c resident; prior processing done
    int cur = c & 1;
    if (c + 1 < NC) {
      size_t go = (size_t)(c + 1) * SCH;
      scan_issue_chunk(ucp + go * DIN, dtp + go * DIN, dblp + go * DBLW,
                       s_uc[cur ^ 1], s_dt[cur ^ 1], s_db[cur ^ 1], t);
    }
#pragma unroll 4
    for (int j = 0; j < SCH; ++j) {
      int l = c * SCH + j;
      float ut  = s_uc[cur][j * DIN + di];
      float dtt = s_dt[cur][j * DIN + di];
      float du  = dtt * ut;
      float y = 0.0f;
#pragma unroll
      for (int i = 0; i < 4; ++i) {
        float Bv = s_db[cur][j * DBLW + DTR + dsq * 4 + i];
        float Cv = s_db[cur][j * DBLW + DTR + DST + dsq * 4 + i];
        h[i] = __expf(dtt * A[i]) * h[i] + du * Bv;
        y += h[i] * Cv;
      }
      y += __shfl_xor(y, 1, 32);
      y += __shfl_xor(y, 2, 32);
      if (dsq == 0) {
        int ol = (d == 0) ? l : (LSEQ - 1 - l);
        float zv = zp[(size_t)ol * DIN + di];
        yout[(size_t)ol * DIN + di] = (y + ut * Dv) * silu_f(zv);
      }
    }
  }
}

// ---------------------------------------------------------------------------
// K7: x_mamba_g = (yf + yb) @ Wout_g^T. K=64. grid(M/16, 2*G), block 32.
// ---------------------------------------------------------------------------
__global__ void k_wout(const float* __restrict__ yf,
                       const float* __restrict__ yb,
                       const float* __restrict__ Wout,
                       float* __restrict__ xm) {
  int lane = threadIdx.x;
  int m0   = blockIdx.x * 16;
  int nt   = blockIdx.y & 1;
  int g    = blockIdx.y >> 1;
  int n0   = nt * 16;

  const float* A0 = yf + ((size_t)g * MTOK + m0) * DIN;
  const float* A1 = yb + ((size_t)g * MTOK + m0) * DIN;
  const float* Wp = Wout + ((size_t)g * DMG + n0) * DIN;

  v8f c = {};
#pragma unroll
  for (int k0 = 0; k0 < DIN; k0 += 32) {
    v16h a = ldA2(A0 + k0, A1 + k0, DIN, lane);
    v16h b = ldBT(Wp + k0, DIN, 16, lane);
    c = WMMA_F32_F16(a, b, c);
  }

  int n = g * DMG + n0 + (lane & 15);
#pragma unroll
  for (int r = 0; r < 8; ++r) {
    int m = m0 + r + 8 * (lane >> 4);
    xm[(size_t)m * CCH + n] = c[r];
  }
}

// ---------------------------------------------------------------------------
// K8: gate = sigmoid(xn @ gate_W^T + b); gated = gate*xm + (1-gate)*xs.
// 256 threads = 8 waves (one n-tile each); A tile converted to f16 in LDS
// once and reused by all 8 waves. grid(M/16).
// ---------------------------------------------------------------------------
__global__ void k_gate(const float* __restrict__ xn,
                       const float* __restrict__ gW,
                       const float* __restrict__ gb,
                       const float* __restrict__ xm,
                       const float* __restrict__ xs,
                       float* __restrict__ gated) {
  __shared__ _Float16 sA[16 * CCH];
  int tid = threadIdx.x;
  int m0  = blockIdx.x * 16;

  for (int i = tid; i < 16 * CCH; i += 256) {
    int m = i >> 7, k = i & (CCH - 1);
    sA[i] = (_Float16)xn[(size_t)(m0 + m) * CCH + k];
  }
  __syncthreads();

  int wave = tid >> 5, lane = tid & 31;
  int n0 = wave * 16;
  const float* Wp = gW + (size_t)n0 * CCH;

  v8f c = {};
#pragma unroll
  for (int k0 = 0; k0 < CCH; k0 += 32) {
    v16h a = ldA_sh(sA, k0, lane);
    v16h b = ldBT(Wp + k0, CCH, 16, lane);
    c = WMMA_F32_F16(a, b, c);
  }

  int n = n0 + (lane & 15);
  float bias = gb[n];
#pragma unroll
  for (int r = 0; r < 8; ++r) {
    int m = m0 + r + 8 * (lane >> 4);
    size_t o = (size_t)m * CCH + n;
    float gv = 1.0f / (1.0f + __expf(-(c[r] + bias)));
    gated[o] = gv * xm[o] + (1.0f - gv) * xs[o];
  }
}

// ---------------------------------------------------------------------------
// K9: out = gated @ proj_W^T + proj_b -> NCHW. Same LDS-shared-A scheme.
// ---------------------------------------------------------------------------
__global__ void k_proj(const float* __restrict__ gated,
                       const float* __restrict__ pW,
                       const float* __restrict__ pb,
                       float* __restrict__ out) {
  __shared__ _Float16 sA[16 * CCH];
  int tid = threadIdx.x;
  int m0  = blockIdx.x * 16;

  for (int i = tid; i < 16 * CCH; i += 256) {
    int m = i >> 7, k = i & (CCH - 1);
    sA[i] = (_Float16)gated[(size_t)(m0 + m) * CCH + k];
  }
  __syncthreads();

  int wave = tid >> 5, lane = tid & 31;
  int n0 = wave * 16;
  const float* Wp = pW + (size_t)n0 * CCH;

  v8f c = {};
#pragma unroll
  for (int k0 = 0; k0 < CCH; k0 += 32) {
    v16h a = ldA_sh(sA, k0, lane);
    v16h b = ldBT(Wp + k0, CCH, 16, lane);
    c = WMMA_F32_F16(a, b, c);
  }

  int n = n0 + (lane & 15);
  float bias = pb[n];
#pragma unroll
  for (int r = 0; r < 8; ++r) {
    int m = m0 + r + 8 * (lane >> 4);
    int b = m >> 12;
    int l = m & (LSEQ - 1);
    out[((size_t)b * NOUT + n) * LSEQ + l] = c[r] + bias;
  }
}

// ---------------------------------------------------------------------------
// Host launch
// ---------------------------------------------------------------------------
extern "C" void kernel_launch(void* const* d_in, const int* in_sizes, int n_in,
                              void* d_out, int out_size, void* d_ws, size_t ws_size,
                              hipStream_t stream) {
  const float* x     = (const float*)d_in[0];
  const float* pcw   = (const float*)d_in[1];
  const float* pcb   = (const float*)d_in[2];
  const float* pemb  = (const float*)d_in[3];
  const float* lng   = (const float*)d_in[4];
  const float* lnb   = (const float*)d_in[5];
  const float* gW    = (const float*)d_in[6];
  const float* gb    = (const float*)d_in[7];
  const float* pW    = (const float*)d_in[8];
  const float* pb    = (const float*)d_in[9];
  const float* Win   = (const float*)d_in[10];
  const float* Wout  = (const float*)d_in[11];
  const float* cw    = (const float*)d_in[12];
  const float* cb    = (const float*)d_in[13];
  const float* xw    = (const float*)d_in[14];
  const float* dtw   = (const float*)d_in[15];
  const float* dtb   = (const float*)d_in[16];
  const float* Alog  = (const float*)d_in[17];
  const float* Dsk   = (const float*)d_in[18];
  float* out = (float*)d_out;

  float* ws = (float*)d_ws;
  size_t off = 0;
  float* xs    = ws + off; off += (size_t)MTOK * CCH;
  float* xn    = ws + off; off += (size_t)MTOK * CCH;
  float* u     = ws + off; off += (size_t)NG * MTOK * DIN;
  float* z     = ws + off; off += (size_t)NG * MTOK * DIN;
  float* uc    = ws + off; off += (size_t)NG * 2 * MTOK * DIN;
  float* dbl   = ws + off; off += (size_t)NG * 2 * MTOK * DBLW;
  float* dt    = ws + off; off += (size_t)NG * 2 * MTOK * DIN;
  float* yf    = ws + off; off += (size_t)NG * MTOK * DIN;
  float* yb    = ws + off; off += (size_t)NG * MTOK * DIN;
  float* xm    = ws + off; off += (size_t)MTOK * CCH;
  float* gated = ws + off; off += (size_t)MTOK * CCH;
  (void)ws_size; (void)in_sizes; (void)n_in; (void)out_size;

  k_prep<<<dim3(MTOK), dim3(CCH), 0, stream>>>(x, pcw, pcb, pemb, lng, lnb, xs, xn);

  k_inproj<<<dim3(MTOK / 16, 8 * NG), dim3(32), 0, stream>>>(xn, Win, u, z);

  {
    size_t n = (size_t)NG * 2 * MTOK * DIN;
    k_dwconv<<<dim3((unsigned)(n / 256)), dim3(256), 0, stream>>>(u, cw, cb, uc);
  }

  k_xproj<<<dim3(MTOK / 16, 3 * 2 * NG), dim3(32), 0, stream>>>(uc, xw, dbl);

  {
    size_t n = (size_t)NG * 2 * MTOK * DIN;
    k_dt<<<dim3((unsigned)(n / 256)), dim3(256), 0, stream>>>(dbl, dtw, dtb, dt);
  }

  k_scan<<<dim3(NG * 2 * BSZ), dim3(256), 0, stream>>>(uc, dt, dbl, z, Alog, Dsk, yf, yb);

  k_wout<<<dim3(MTOK / 16, 2 * NG), dim3(32), 0, stream>>>(yf, yb, Wout, xm);

  k_gate<<<dim3(MTOK / 16), dim3(256), 0, stream>>>(xn, gW, gb, xm, xs, gated);

  k_proj<<<dim3(MTOK / 16), dim3(256), 0, stream>>>(gated, pW, pb, out);
}
